// Qwen2MoeSparseMoeBlock_23089744183632
// MI455X (gfx1250) — matrix operand order repaired and checked by reading, not compiled, not run on hardware
//
#include <hip/hip_runtime.h>
#include <stdint.h>

#define N_TOK 1024
#define HDIM  2048
#define NEXP  8
#define IDIM  1408
#define ISDIM 5632

typedef __bf16 bf16_t;
typedef bf16_t v16bf __attribute__((ext_vector_type(16)));
typedef bf16_t v2bf  __attribute__((ext_vector_type(2)));
typedef float  v8f   __attribute__((ext_vector_type(8)));

#define SA 40  // LDS row stride in ushort elements (32 data + 8 pad -> 80B rows, 16B aligned)

union Frag { v16bf bf; uint4 q[2]; };

// single f32 -> bf16 (native convert, RNE)
__device__ __forceinline__ unsigned short f2bf1(float x) {
  __bf16 b = (__bf16)x;
  return __builtin_bit_cast(unsigned short, b);
}
// pack two f32 -> 2xbf16 in one dword (lo in [15:0]); lowers to v_cvt_pk_bf16_f32
__device__ __forceinline__ unsigned int pack2(float lo, float hi) {
  v2bf v;
  v.x = (__bf16)lo;
  v.y = (__bf16)hi;
  return __builtin_bit_cast(unsigned int, v);
}

__device__ __forceinline__ float silu_f(float x) { return x / (1.0f + __expf(-x)); }

// A fragment 16x32 bf16: lane half selects K offset 8 within each 16-block (ISA 7.12.2)
__device__ __forceinline__ Frag ld_frag_a(const unsigned short* base, int lane, int row0) {
  int m = row0 + (lane & 15);
  int half = (lane >> 4) & 1;
  const unsigned short* p = base + m * SA + half * 8;
  Frag f;
  f.q[0] = *(const uint4*)(p);        // K = half*8 .. +7
  f.q[1] = *(const uint4*)(p + 16);   // K = 16+half*8 .. +7
  return f;
}

// B fragment 32x16 bf16 from N-major (transposed) LDS tile: lane half selects K offset 8 within each 16-block (matches A)
__device__ __forceinline__ Frag ld_frag_b(const unsigned short* base, int lane, int col0) {
  int n = col0 + (lane & 15);
  int half = (lane >> 4) & 1;
  const unsigned short* p = base + n * SA + half * 8;
  Frag f;
  f.q[0] = *(const uint4*)(p);
  f.q[1] = *(const uint4*)(p + 16);
  return f;
}

// ---------------- GEMM1: act = silu(X@Wg) * (X@Wu), output bf16 ----------------
__device__ __forceinline__ void gemm1_body(
    const float* __restrict__ X, const int* __restrict__ gather,
    int cnt, int m0, int c0,
    const float* __restrict__ W, int ldb, int upOff,
    unsigned short* __restrict__ act, int ldact, int aRowBase,
    unsigned short* As, unsigned short* Bg, unsigned short* Bu)
{
  const int tid  = threadIdx.x;
  const int lane = tid & 31;
  const int wave = tid >> 5;
  const int wm   = wave & 3;   // 4 M-groups
  const int wn   = wave >> 2;  // 2 N-groups

  v8f aG0 = {0,0,0,0,0,0,0,0}, aG1 = aG0, aU0 = aG0, aU1 = aG0;

  // A staging: thread -> (row rA, 8 consecutive K). Clamp row (garbage rows are masked in epilogue).
  const int rA = tid >> 2;            // 0..63
  const int cA = (tid & 3) << 3;      // 0,8,16,24
  int rowA = m0 + rA;
  if (rowA >= cnt) rowA = cnt - 1;
  const int tokA = gather ? gather[rowA] : rowA;
  const float* srcA = X + (size_t)tokA * HDIM + cA;

  // B staging: waves 0-3 stage gate, waves 4-7 stage up. Each thread: K-pair x 8 N.
  const int tb = tid & 127;
  const int k2 = tb >> 3;             // 0..15 -> k = 2*k2
  const int nB = (tb & 7) << 3;       // 0..56
  unsigned short* BT = (tid < 128) ? Bg : Bu;
  const float* srcB = W + (size_t)(2 * k2) * ldb + c0 + nB + ((tid < 128) ? 0 : upOff);

  // prefetch k0 = 0
  float4 pa0 = ((const float4*)srcA)[0];
  float4 pa1 = ((const float4*)srcA)[1];
  float4 pb00 = ((const float4*)srcB)[0];
  float4 pb01 = ((const float4*)srcB)[1];
  float4 pb10 = ((const float4*)(srcB + ldb))[0];
  float4 pb11 = ((const float4*)(srcB + ldb))[1];

  for (int k0 = 0; k0 < HDIM; k0 += 32) {
    __syncthreads();
    { // store A (pack to bf16)
      union { unsigned int u[4]; uint4 q; } t;
      t.u[0] = pack2(pa0.x, pa0.y);
      t.u[1] = pack2(pa0.z, pa0.w);
      t.u[2] = pack2(pa1.x, pa1.y);
      t.u[3] = pack2(pa1.z, pa1.w);
      *(uint4*)&As[rA * SA + cA] = t.q;
    }
    { // store B transposed: dword = bf16(k), bf16(k+1) at row n
      float r0[8] = {pb00.x, pb00.y, pb00.z, pb00.w, pb01.x, pb01.y, pb01.z, pb01.w};
      float r1[8] = {pb10.x, pb10.y, pb10.z, pb10.w, pb11.x, pb11.y, pb11.z, pb11.w};
#pragma unroll
      for (int j = 0; j < 8; ++j)
        *(unsigned int*)&BT[(nB + j) * SA + 2 * k2] = pack2(r0[j], r1[j]);
    }
    // prefetch next K-step (overlaps with this iteration's WMMA)
    if (k0 + 32 < HDIM) {
      const float* nA = srcA + k0 + 32;
      pa0 = ((const float4*)nA)[0];
      pa1 = ((const float4*)nA)[1];
      const float* nBp = srcB + (size_t)(k0 + 32) * ldb;
      pb00 = ((const float4*)nBp)[0];
      pb01 = ((const float4*)nBp)[1];
      pb10 = ((const float4*)(nBp + ldb))[0];
      pb11 = ((const float4*)(nBp + ldb))[1];
    }
    __syncthreads();
    Frag a   = ld_frag_a(As, lane, wm * 16);
    Frag bg0 = ld_frag_b(Bg, lane, wn * 32);
    Frag bg1 = ld_frag_b(Bg, lane, wn * 32 + 16);
    Frag bu0 = ld_frag_b(Bu, lane, wn * 32);
    Frag bu1 = ld_frag_b(Bu, lane, wn * 32 + 16);
    aG0 = __builtin_amdgcn_wmma_f32_16x16x32_bf16(false, a.bf, false, bg0.bf, (short)0, aG0, false, false);
    aG1 = __builtin_amdgcn_wmma_f32_16x16x32_bf16(false, a.bf, false, bg1.bf, (short)0, aG1, false, false);
    aU0 = __builtin_amdgcn_wmma_f32_16x16x32_bf16(false, a.bf, false, bu0.bf, (short)0, aU0, false, false);
    aU1 = __builtin_amdgcn_wmma_f32_16x16x32_bf16(false, a.bf, false, bu1.bf, (short)0, aU1, false, false);
  }

  const int nn = lane & 15;
  const int hi = lane >> 4;
#pragma unroll
  for (int v = 0; v < 8; ++v) {
    int m = wm * 16 + v + 8 * hi;   // C/D layout: VGPR v -> M = v + 8*(lane>>4)
    int row = m0 + m;
    if (row < cnt) {
      size_t arow = (size_t)(aRowBase + row);
      act[arow * ldact + c0 + wn * 32 + nn]      = f2bf1(silu_f(aG0[v]) * aU0[v]);
      act[arow * ldact + c0 + wn * 32 + 16 + nn] = f2bf1(silu_f(aG1[v]) * aU1[v]);
    }
  }
}

// ---------------- GEMM2: Out = (act_bf16 @ W) * rowScale, BN = 128 ----------------
__device__ __forceinline__ void gemm2_body(
    const unsigned short* __restrict__ act, int aRowBase, int cnt, int m0, int c0, int Kdim,
    const float* __restrict__ W, float* __restrict__ Out,
    const float* __restrict__ rowScale,
    unsigned short* As, unsigned short* Bs /* 128*SA */)
{
  const int tid  = threadIdx.x;
  const int lane = tid & 31;
  const int wave = tid >> 5;
  const int wm   = wave & 3;
  const int wn   = wave >> 2;  // 2 groups x 64 cols

  v8f a0 = {0,0,0,0,0,0,0,0}, a1 = a0, a2 = a0, a3 = a0;

  // A tile: pure bf16 copy global->LDS via async DMA (ASYNCcnt)
  const int rA = tid >> 2;
  const int cA = (tid & 3) << 3;
  int rowA = m0 + rA;
  if (rowA >= cnt) rowA = cnt - 1;
  const unsigned short* srcA = act + (size_t)(aRowBase + rowA) * Kdim + cA;
  unsigned int ldsA = (unsigned int)(size_t)&As[rA * SA + cA];

  // B staging: K-pair x 8 N per thread over 32x128 tile
  const int k2 = tid >> 4;            // 0..15
  const int nB = (tid & 15) << 3;     // 0..120
  const float* srcB = W + (size_t)(2 * k2) * HDIM + c0 + nB;

  float4 pb00 = ((const float4*)srcB)[0];
  float4 pb01 = ((const float4*)srcB)[1];
  float4 pb10 = ((const float4*)(srcB + HDIM))[0];
  float4 pb11 = ((const float4*)(srcB + HDIM))[1];

  for (int k0 = 0; k0 < Kdim; k0 += 32) {
    __syncthreads();
    // async A copy: 16 bytes per lane, 256 lanes -> 64x32 bf16 tile
    asm volatile("global_load_async_to_lds_b128 %0, %1, off"
                 :: "v"(ldsA), "v"(srcA + k0)
                 : "memory");
    { // store B transposed pairs
      float r0[8] = {pb00.x, pb00.y, pb00.z, pb00.w, pb01.x, pb01.y, pb01.z, pb01.w};
      float r1[8] = {pb10.x, pb10.y, pb10.z, pb10.w, pb11.x, pb11.y, pb11.z, pb11.w};
#pragma unroll
      for (int j = 0; j < 8; ++j)
        *(unsigned int*)&Bs[(nB + j) * SA + 2 * k2] = pack2(r0[j], r1[j]);
    }
    if (k0 + 32 < Kdim) {
      const float* nBp = srcB + (size_t)(k0 + 32) * HDIM;
      pb00 = ((const float4*)nBp)[0];
      pb01 = ((const float4*)nBp)[1];
      pb10 = ((const float4*)(nBp + HDIM))[0];
      pb11 = ((const float4*)(nBp + HDIM))[1];
    }
    asm volatile("s_wait_asynccnt 0" ::: "memory");
    __syncthreads();
    Frag a  = ld_frag_a(As, lane, wm * 16);
    Frag b0 = ld_frag_b(Bs, lane, wn * 64);
    Frag b1 = ld_frag_b(Bs, lane, wn * 64 + 16);
    Frag b2 = ld_frag_b(Bs, lane, wn * 64 + 32);
    Frag b3 = ld_frag_b(Bs, lane, wn * 64 + 48);
    a0 = __builtin_amdgcn_wmma_f32_16x16x32_bf16(false, a.bf, false, b0.bf, (short)0, a0, false, false);
    a1 = __builtin_amdgcn_wmma_f32_16x16x32_bf16(false, a.bf, false, b1.bf, (short)0, a1, false, false);
    a2 = __builtin_amdgcn_wmma_f32_16x16x32_bf16(false, a.bf, false, b2.bf, (short)0, a2, false, false);
    a3 = __builtin_amdgcn_wmma_f32_16x16x32_bf16(false, a.bf, false, b3.bf, (short)0, a3, false, false);
  }

  const int nn = lane & 15;
  const int hi = lane >> 4;
#pragma unroll
  for (int v = 0; v < 8; ++v) {
    int m = wm * 16 + v + 8 * hi;
    int row = m0 + m;
    if (row < cnt) {
      float sc = rowScale ? rowScale[row] : 1.0f;
      size_t orow = (size_t)(aRowBase + row);
      float* op = Out + orow * HDIM + c0 + wn * 64 + nn;
      op[0]  = a0[v] * sc;
      op[16] = a1[v] * sc;
      op[32] = a2[v] * sc;
      op[48] = a3[v] * sc;
    }
  }
}

// ---------------- kernels ----------------
__global__ void k_zero_counts(int* counts) {
  if (threadIdx.x < NEXP) counts[threadIdx.x] = 0;
}

__global__ __launch_bounds__(256) void k_router(
    const float* __restrict__ x, const float* __restrict__ gw,
    const float* __restrict__ sgw,
    int* __restrict__ counts, int* __restrict__ toks,
    int* __restrict__ texp, int* __restrict__ tslot, float* __restrict__ tw,
    float* __restrict__ sg)
{
  int lane = threadIdx.x & 31;
  int token = blockIdx.x * 8 + (threadIdx.x >> 5);
  const float* xr = x + (size_t)token * HDIM;
  float acc[8] = {0,0,0,0,0,0,0,0};
  float accs = 0.0f;
  for (int h = lane; h < HDIM; h += 32) {
    float xv = xr[h];
    const float4* g = (const float4*)(gw + (size_t)h * NEXP);
    float4 g0 = g[0], g1 = g[1];
    acc[0] += xv * g0.x; acc[1] += xv * g0.y; acc[2] += xv * g0.z; acc[3] += xv * g0.w;
    acc[4] += xv * g1.x; acc[5] += xv * g1.y; acc[6] += xv * g1.z; acc[7] += xv * g1.w;
    accs += xv * sgw[h];
  }
#pragma unroll
  for (int off = 16; off > 0; off >>= 1) {
#pragma unroll
    for (int e = 0; e < 8; ++e) acc[e] += __shfl_down(acc[e], off);
    accs += __shfl_down(accs, off);
  }
  if (lane == 0) {
    float mx = acc[0];
#pragma unroll
    for (int e = 1; e < 8; ++e) mx = fmaxf(mx, acc[e]);
    float p[8]; float s = 0.0f;
#pragma unroll
    for (int e = 0; e < 8; ++e) { p[e] = __expf(acc[e] - mx); s += p[e]; }
    float inv = 1.0f / s;
    int i1 = 0;
#pragma unroll
    for (int e = 1; e < 8; ++e) if (p[e] > p[i1]) i1 = e;
    int i2 = (i1 == 0) ? 1 : 0;
#pragma unroll
    for (int e = 0; e < 8; ++e) if (e != i1 && p[e] > p[i2]) i2 = e;
    int s1 = atomicAdd(&counts[i1], 1);
    int s2 = atomicAdd(&counts[i2], 1);
    toks[i1 * N_TOK + s1] = token;
    toks[i2 * N_TOK + s2] = token;
    texp[2*token] = i1;  texp[2*token+1] = i2;
    tslot[2*token] = s1; tslot[2*token+1] = s2;
    tw[2*token] = p[i1] * inv; tw[2*token+1] = p[i2] * inv;
    sg[token] = 1.0f / (1.0f + __expf(-accs));
  }
}

__global__ void k_prefix(const int* counts, int* offsets) {
  if (threadIdx.x == 0) {
    int s = 0;
    for (int e = 0; e < NEXP; ++e) { offsets[e] = s; s += counts[e]; }
    offsets[NEXP] = s;
  }
}

__global__ __launch_bounds__(256) void k_expert_gemm1(
    const float* __restrict__ x, const float* __restrict__ w_gu,
    const int* __restrict__ counts, const int* __restrict__ offsets,
    const int* __restrict__ toks, unsigned short* __restrict__ act)
{
  __shared__ unsigned short As[64*SA], Bg[64*SA], Bu[64*SA];
  int e = blockIdx.z;
  int cnt = counts[e];
  int m0 = blockIdx.y * 64;
  if (m0 >= cnt) return;
  gemm1_body(x, toks + e * N_TOK, cnt, m0, blockIdx.x * 64,
             w_gu + (size_t)e * HDIM * (2*IDIM), 2*IDIM, IDIM,
             act, IDIM, offsets[e], As, Bg, Bu);
}

__global__ __launch_bounds__(256) void k_shared_gemm1(
    const float* __restrict__ x, const float* __restrict__ w_gu,
    unsigned short* __restrict__ act)
{
  __shared__ unsigned short As[64*SA], Bg[64*SA], Bu[64*SA];
  gemm1_body(x, nullptr, N_TOK, blockIdx.y * 64, blockIdx.x * 64,
             w_gu, 2*ISDIM, ISDIM, act, ISDIM, 0, As, Bg, Bu);
}

__global__ __launch_bounds__(256) void k_expert_gemm2(
    const unsigned short* __restrict__ act, const float* __restrict__ w_down,
    const int* __restrict__ counts, const int* __restrict__ offsets,
    float* __restrict__ eo)
{
  __shared__ unsigned short As[64*SA], Bs[128*SA];
  int e = blockIdx.z;
  int cnt = counts[e];
  int m0 = blockIdx.y * 64;
  if (m0 >= cnt) return;
  gemm2_body(act, offsets[e], cnt, m0, blockIdx.x * 128, IDIM,
             w_down + (size_t)e * IDIM * HDIM, eo, nullptr, As, Bs);
}

__global__ __launch_bounds__(256) void k_shared_gemm2(
    const unsigned short* __restrict__ act, const float* __restrict__ w_down,
    const float* __restrict__ sg, float* __restrict__ out)
{
  __shared__ unsigned short As[64*SA], Bs[128*SA];
  gemm2_body(act, 0, N_TOK, blockIdx.y * 64, blockIdx.x * 128, ISDIM,
             w_down, out, sg, As, Bs);
}

__global__ __launch_bounds__(256) void k_combine(
    float* __restrict__ out, const float* __restrict__ eo,
    const int* __restrict__ offsets, const int* __restrict__ texp,
    const int* __restrict__ tslot, const float* __restrict__ tw)
{
  int t = blockIdx.x;
  int e1 = texp[2*t], e2 = texp[2*t+1];
  size_t r1 = (size_t)(offsets[e1] + tslot[2*t]);
  size_t r2 = (size_t)(offsets[e2] + tslot[2*t+1]);
  float w1 = tw[2*t], w2 = tw[2*t+1];
  const float4* p1 = (const float4*)(eo + r1 * HDIM) + threadIdx.x * 2;
  const float4* p2 = (const float4*)(eo + r2 * HDIM) + threadIdx.x * 2;
  float4* po = (float4*)(out + (size_t)t * HDIM) + threadIdx.x * 2;
#pragma unroll
  for (int j = 0; j < 2; ++j) {
    float4 o = po[j], a = p1[j], b = p2[j];
    o.x += w1 * a.x + w2 * b.x;
    o.y += w1 * a.y + w2 * b.y;
    o.z += w1 * a.z + w2 * b.z;
    o.w += w1 * a.w + w2 * b.w;
    po[j] = o;
  }
}

extern "C" void kernel_launch(void* const* d_in, const int* in_sizes, int n_in,
                              void* d_out, int out_size, void* d_ws, size_t ws_size,
                              hipStream_t stream)
{
  const float* x   = (const float*)d_in[0];
  const float* gw  = (const float*)d_in[1];
  const float* wgu = (const float*)d_in[2];
  const float* wdn = (const float*)d_in[3];
  const float* sgu = (const float*)d_in[4];
  const float* sdn = (const float*)d_in[5];
  const float* sgw = (const float*)d_in[6];
  float* out = (float*)d_out;

  char* ws = (char*)d_ws;
  int*            counts  = (int*)(ws + 0);          // 8 ints
  int*            offsets = (int*)(ws + 64);         // 9 ints
  float*          sg      = (float*)(ws + 128);      // N floats
  int*            texp    = (int*)(ws + 4224);       // 2N ints
  int*            tslot   = (int*)(ws + 12416);      // 2N ints
  float*          tw      = (float*)(ws + 20608);    // 2N floats
  int*            toks    = (int*)(ws + 28800);      // E*N ints
  unsigned short* acte    = (unsigned short*)(ws + 61696);    // 2048 x IDIM bf16
  float*          eo      = (float*)(ws + 5828864);           // 2048 x HDIM f32
  unsigned short* acts    = (unsigned short*)(ws + 22606080); // N x ISDIM bf16
  // total ws use ~34.1 MB

  k_zero_counts<<<1, 32, 0, stream>>>(counts);
  k_router<<<N_TOK / 8, 256, 0, stream>>>(x, gw, sgw, counts, toks, texp, tslot, tw, sg);
  k_prefix<<<1, 1, 0, stream>>>(counts, offsets);
  k_expert_gemm1<<<dim3(IDIM / 64, N_TOK / 64, NEXP), 256, 0, stream>>>(x, wgu, counts, offsets, toks, acte);
  k_shared_gemm1<<<dim3(ISDIM / 64, N_TOK / 64), 256, 0, stream>>>(x, sgu, acts);
  k_shared_gemm2<<<dim3(HDIM / 128, N_TOK / 64), 256, 0, stream>>>(acts, sdn, sg, out);
  k_expert_gemm2<<<dim3(HDIM / 128, N_TOK / 64, NEXP), 256, 0, stream>>>(acte, wdn, counts, offsets, eo);
  k_combine<<<N_TOK, 256, 0, stream>>>(out, eo, offsets, texp, tslot, tw);
}
